// PrimalGNN_25546465477048
// MI455X (gfx1250) — matrix operand order repaired and checked by reading, not compiled, not run on hardware
//
#include <hip/hip_runtime.h>
#include <hip/hip_bf16.h>

// ---------------- problem constants (match reference) ----------------
#define BB   64
#define LL   2048
#define GG   512
#define FF   3072
#define NN   (LL + GG + FF)      // 5632 nodes per graph
#define NT   (BB * NN)           // 360448 total nodes
#define EG_  16384
#define EE   (BB * EG_)          // 1048576 edges
#define HH   64
#define DIN  7
#define NLAYER 4
#define EPS  1e-5f

typedef __attribute__((ext_vector_type(16))) _Float16 v16h;
typedef __attribute__((ext_vector_type(8)))  float    v8f;

// ---------------- encoder: h = relu(x @ enc_w.T + enc_b) ----------------
__global__ void enc_kernel(const float* __restrict__ x,
                           const float* __restrict__ enc_w,
                           const float* __restrict__ enc_b,
                           float* __restrict__ h) {
    int tid = blockIdx.x * blockDim.x + threadIdx.x;   // over NT*64
    if (tid >= NT * HH) return;
    int n = tid >> 6, j = tid & 63;
    float acc = enc_b[j];
#pragma unroll
    for (int d = 0; d < DIN; ++d)
        acc += x[n * DIN + d] * enc_w[j * DIN + d];
    h[tid] = fmaxf(acc, 0.0f);
}

// ---------------- LayerNorm: one wave32 per node (64 feats = 2/lane) ----------------
__global__ void ln_kernel(const float* __restrict__ h,
                          const float* __restrict__ g,
                          const float* __restrict__ b,
                          float* __restrict__ hn) {
    int wid  = (blockIdx.x * blockDim.x + threadIdx.x) >> 5;  // node id
    int lane = threadIdx.x & 31;
    if (wid >= NT) return;
    const float* row = h + (size_t)wid * HH;
    float v0 = row[lane], v1 = row[lane + 32];
    float s = v0 + v1, q = v0 * v0 + v1 * v1;
#pragma unroll
    for (int o = 16; o > 0; o >>= 1) {
        s += __shfl_xor(s, o, 32);
        q += __shfl_xor(q, o, 32);
    }
    float mu  = s * (1.0f / HH);
    float var = q * (1.0f / HH) - mu * mu;
    float rs  = rsqrtf(var + EPS);
    float* out = hn + (size_t)wid * HH;
    out[lane]      = (v0 - mu) * rs * g[lane]      + b[lane];
    out[lane + 32] = (v1 - mu) * rs * g[lane + 32] + b[lane + 32];
}

// ---------------- scatter-add: agg[dst] += hn[src], native f32 atomics ----------------
__global__ void scatter_kernel(const float* __restrict__ hn,
                               const int* __restrict__ src,
                               const int* __restrict__ dst,
                               float* __restrict__ agg) {
    int tid = blockIdx.x * blockDim.x + threadIdx.x;   // over EE*64 (= 2^26, fits int)
    int e = tid >> 6, f = tid & 63;
    int s = src[e], d = dst[e];
    unsafeAtomicAdd(&agg[(size_t)d * HH + f], hn[(size_t)s * HH + f]);
}

// ------- GraphConv GEMM: h = relu(agg @ w_rel^T + b_rel + hn @ w_root^T) -------
// One wave per 16-row node tile; WMMA f32<-f16 16x16x32, K=64 in 2 steps,
// 4 N-tiles x (rel + root) = 16 v_wmma per wave. Layouts per CDNA5 ISA 7.12.2.
__global__ void __launch_bounds__(256)
gconv_kernel(const float* __restrict__ agg,
             const float* __restrict__ hn,
             const float* __restrict__ w_rel,   // [64,64] row-major
             const float* __restrict__ b_rel,   // [64]
             const float* __restrict__ w_root,  // [64,64]
             float* __restrict__ hout) {
    const int lane = threadIdx.x & 31;
    const int wave = threadIdx.x >> 5;
    const int tile = blockIdx.x * 8 + wave;          // NT/16 = 22528 tiles exactly
    const int m  = lane & 15;
    const int hi = lane >> 4;
    const size_t rowbase = (size_t)(tile * 16 + m) * HH;

    // A operands (16x32 f16): VGPR v holds K = (v/4)*16 + hi*8 + (v%4)*2 (+ks*32)
    v16h a_rel[2], a_root[2];
#pragma unroll
    for (int ks = 0; ks < 2; ++ks) {
#pragma unroll
        for (int v = 0; v < 8; ++v) {
            int kb = ks * 32 + (v >> 2) * 16 + hi * 8 + (v & 3) * 2;
            a_rel[ks][2 * v]      = (_Float16)agg[rowbase + kb];
            a_rel[ks][2 * v + 1]  = (_Float16)agg[rowbase + kb + 1];
            a_root[ks][2 * v]     = (_Float16)hn[rowbase + kb];
            a_root[ks][2 * v + 1] = (_Float16)hn[rowbase + kb + 1];
        }
    }

#pragma unroll
    for (int jt = 0; jt < 4; ++jt) {
        const int n = jt * 16 + m;                   // output column for this lane
        v8f acc = {};
#pragma unroll
        for (int ks = 0; ks < 2; ++ks) {
            // B (32x16 f16) = W^T: lane holds K = (i/8)*16 + hi*8 + i%8 of row n of W
            const int koff = ks * 32 + hi * 8;
            v16h brel, broot;
#pragma unroll
            for (int i = 0; i < 16; ++i) {
                brel[i]  = (_Float16)w_rel[n * HH + koff + (i >> 3) * 16 + (i & 7)];
                broot[i] = (_Float16)w_root[n * HH + koff + (i >> 3) * 16 + (i & 7)];
            }
            acc = __builtin_amdgcn_wmma_f32_16x16x32_f16(
                      false, a_rel[ks], false, brel, (short)0, acc, false, false);
            acc = __builtin_amdgcn_wmma_f32_16x16x32_f16(
                      false, a_root[ks], false, broot, (short)0, acc, false, false);
        }
        const float bias = b_rel[n];
#pragma unroll
        for (int r = 0; r < 8; ++r) {                // D VGPR r -> row r + 8*hi
            float v = acc[r] + bias;
            hout[(size_t)(tile * 16 + r + 8 * hi) * HH + n] = v > 0.0f ? v : 0.0f;
        }
    }
}

// ---------------- linear heads: out[i] = h[idx[i]] . w + b ----------------
__global__ void head_kernel(const float* __restrict__ h,
                            const int* __restrict__ idx,
                            const float* __restrict__ w,
                            const float* __restrict__ bptr,
                            float* __restrict__ out, int count) {
    int i = blockIdx.x * blockDim.x + threadIdx.x;
    if (i >= count) return;
    size_t node = (size_t)idx[i];
    float acc = bptr[0];
#pragma unroll
    for (int j = 0; j < HH; ++j)
        acc += h[node * HH + j] * w[j];
    out[i] = acc;
}

// ---- md[b,l] = demand[b,l] - (p[b] . gen_mask[l] + f[b] . line_mask[l]) ----
// lane layout: b = tid & 63 (B==64), l = tid >> 6 -> mask reads wave-uniform.
__global__ void md_kernel(const float* __restrict__ x,
                          const int* __restrict__ loc_idx,
                          const float* __restrict__ p,
                          const float* __restrict__ f,
                          const float* __restrict__ gm,   // [L,G]
                          const float* __restrict__ lm,   // [L,F]
                          float* __restrict__ md) {
    int tid = blockIdx.x * blockDim.x + threadIdx.x;   // over B*L
    if (tid >= BB * LL) return;
    int b = tid & 63, l = tid >> 6;
    float acc = 0.0f;
    const float* gr = gm + (size_t)l * GG;
    const float* lr = lm + (size_t)l * FF;
    const float* pb = p + (size_t)b * GG;
    const float* fb = f + (size_t)b * FF;
    for (int gI = 0; gI < GG; ++gI) acc += pb[gI] * gr[gI];
    for (int k = 0; k < FF; ++k)    acc += fb[k]  * lr[k];
    float demand = x[(size_t)loc_idx[(size_t)b * LL + l] * DIN + 0];
    md[(size_t)b * LL + l] = demand - acc;
}

extern "C" void kernel_launch(void* const* d_in, const int* in_sizes, int n_in,
                              void* d_out, int out_size, void* d_ws, size_t ws_size,
                              hipStream_t stream) {
    const float* x        = (const float*)d_in[0];
    const int*   edge     = (const int*)  d_in[1];   // [2, E] : src then dst
    const int*   prod_idx = (const int*)  d_in[2];
    const int*   line_idx = (const int*)  d_in[3];
    const int*   loc_idx  = (const int*)  d_in[4];
    const float* enc_w    = (const float*)d_in[5];
    const float* enc_b    = (const float*)d_in[6];
    const float* ln_g     = (const float*)d_in[7];
    const float* ln_b     = (const float*)d_in[8];
    const float* w_rel    = (const float*)d_in[9];   // [4,64,64]
    const float* b_rel    = (const float*)d_in[10];  // [4,64]
    const float* w_root   = (const float*)d_in[11];  // [4,64,64]
    const float* prod_w   = (const float*)d_in[12];
    const float* prod_b   = (const float*)d_in[13];
    const float* flow_w   = (const float*)d_in[14];
    const float* flow_b   = (const float*)d_in[15];
    const float* gen_mask = (const float*)d_in[16];
    const float* line_mask= (const float*)d_in[17];

    const int* src = edge;
    const int* dst = edge + EE;

    // workspace: h, hn, agg  (each NT*64 f32 = 92.3 MB)
    const size_t feat = (size_t)NT * HH;
    float* h   = (float*)d_ws;
    float* hn  = h  + feat;
    float* agg = hn + feat;

    float* out_p  = (float*)d_out;               // [B*G]
    float* out_f  = out_p + (size_t)BB * GG;     // [B*F]
    float* out_md = out_f + (size_t)BB * FF;     // [B*L]

    // encoder
    enc_kernel<<<(NT * HH + 255) / 256, 256, 0, stream>>>(x, enc_w, enc_b, h);

    for (int layer = 0; layer < NLAYER; ++layer) {
        ln_kernel<<<(NT + 7) / 8, 256, 0, stream>>>(h, ln_g, ln_b, hn);
        hipMemsetAsync(agg, 0, feat * sizeof(float), stream);
        scatter_kernel<<<(EE * HH) / 256, 256, 0, stream>>>(hn, src, dst, agg);
        gconv_kernel<<<NT / 16 / 8, 256, 0, stream>>>(
            agg, hn,
            w_rel  + (size_t)layer * HH * HH,
            b_rel  + (size_t)layer * HH,
            w_root + (size_t)layer * HH * HH,
            h);
    }

    head_kernel<<<(BB * GG + 255) / 256, 256, 0, stream>>>(h, prod_idx, prod_w, prod_b, out_p, BB * GG);
    head_kernel<<<(BB * FF + 255) / 256, 256, 0, stream>>>(h, line_idx, flow_w, flow_b, out_f, BB * FF);
    md_kernel  <<<(BB * LL + 255) / 256, 256, 0, stream>>>(x, loc_idx, out_p, out_f, gen_mask, line_mask, out_md);
}
